// MLA_MODEL_77395310674148
// MI455X (gfx1250) — matrix operand order repaired and checked by reading, not compiled, not run on hardware
//
#include <hip/hip_runtime.h>
#include <stdint.h>

// ---------------------------------------------------------------------------
// MLA projection: cq = rmsnorm(x@wq_a); q = cq@wq_b (+rope on rope dims);
// kv = x@wkv; ckv = rmsnorm(kv[:,:512]); k_rope = rope(kv[:,512:]).
// GEMMs run as bf16 WMMA (v_wmma_f32_16x16x32_bf16) with fp32 accumulate,
// software-pipelined global->reg->LDS staging. f32->bf16 conversion uses
// round-half-up + v_perm_b32 packing (cheap VALU).
// ---------------------------------------------------------------------------

#define N_HEADS 128
#define QK_NOPE 128
#define QK_ROPE 64
#define KV_RANK 512
#define CQ_DIM  1536
#define Q_COLS  (N_HEADS * (QK_NOPE + QK_ROPE))   // 24576
#define KV_COLS (KV_RANK + QK_ROPE)               // 576
#define OUT_C   (Q_COLS + KV_RANK + QK_ROPE)      // 25152
#define EPSF    1e-6f

typedef __bf16 bf16x16 __attribute__((ext_vector_type(16)));
typedef float  floatx8 __attribute__((ext_vector_type(8)));

union FragU {
    bf16x16 v;
    uint4   u[2];
};

// Cheap f32->bf16: round-half-up then take high 16 bits.
__device__ __forceinline__ unsigned short f2bf_fast(float f) {
    union { float f; unsigned u; } x;
    x.f = f;
    return (unsigned short)((x.u + 0x8000u) >> 16);
}

// Pack two f32 into one dword of two bf16 (lo, hi) with a single v_perm_b32.
__device__ __forceinline__ unsigned pack_bf16x2(float lo, float hi) {
    union { float f; unsigned u; } a, b;
    a.f = lo; b.f = hi;
    unsigned au = a.u + 0x8000u;
    unsigned bu = b.u + 0x8000u;
    // dst = { bu[31:16], au[31:16] }
    return __builtin_amdgcn_perm(bu, au, 0x07060302u);
}

template <typename TA> struct VecOf;
template <> struct VecOf<float>          { typedef float4  T; };
template <> struct VecOf<unsigned short> { typedef ushort4 T; };

// Convert a staged 4-vector and deposit 8 bytes into LDS.
__device__ __forceinline__ void stA(unsigned short* dst, const float4& v) {
    uint2 p;
    p.x = pack_bf16x2(v.x, v.y);
    p.y = pack_bf16x2(v.z, v.w);
    *(uint2*)dst = p;
}
__device__ __forceinline__ void stA(unsigned short* dst, const ushort4& v) {
    *(ushort4*)dst = v;
}

// ---------------------------------------------------------------------------
// bf16 WMMA GEMM: C[M,N](fp32, ldc) = A[M,K] * B[K,N].
// A: fp32 or bf16-payload ushort; B: fp32 row-major (converted on the fly).
// Block = 256 threads (8 waves), tile 128x128, BK=32.
// Wave (wr,wc) owns a 64x32 patch = 4x2 WMMA 16x16 accumulators.
// Pipeline: tile k+1 global->regs issued before tile k's WMMAs.
// ---------------------------------------------------------------------------
#define BM 128
#define BN 128
#define BK 32
#define LDS_STRIDE 40   // halves; multiple of 8 -> 16B-aligned frag loads

template <typename TA, bool FULL>
__global__ __launch_bounds__(256)
void gemm_bf16_wmma(const TA* __restrict__ A, const float* __restrict__ B,
                    float* __restrict__ C, int M, int N, int K, int ldc)
{
    __shared__ unsigned short As[BM][LDS_STRIDE];   // [m][k]
    __shared__ unsigned short Bs[BN][LDS_STRIDE];   // [n][k]  (B staged transposed)

    const int tid  = threadIdx.x;
    const int lane = tid & 31;
    const int wave = tid >> 5;
    const int wr   = wave >> 2;        // 0..1 : wave row  (64 rows each)
    const int wc   = wave & 3;         // 0..3 : wave col  (32 cols each)
    const int m0   = blockIdx.y * BM;
    const int n0   = blockIdx.x * BN;

    const int lmod     = lane & 15;
    const int laneHalf = lane >> 4;    // 0 or 1
    const int kA = laneHalf * 8;       // A frag: K 0..7|16..23  vs  8..15|24..31
    const int kB = laneHalf * 8;       // B frag: K 0..7|16..23  vs  8..15|24..31

    // Per-thread staging coordinates (4 groups of 4 elements for each tile).
    int aRow[4], aC4[4], bKr[4], bC4[4];
#pragma unroll
    for (int it = 0; it < 4; ++it) {
        int g = tid + it * 256;
        aRow[it] = g >> 3;            // A: 8 groups per 32-wide row
        aC4[it]  = (g & 7) * 4;
        bKr[it]  = g >> 5;            // B: 32 groups per 128-wide k-row
        bC4[it]  = (g & 31) * 4;
    }

    floatx8 acc[4][2];
#pragma unroll
    for (int mt = 0; mt < 4; ++mt)
#pragma unroll
        for (int nt = 0; nt < 2; ++nt)
#pragma unroll
            for (int g = 0; g < 8; ++g) acc[mt][nt][g] = 0.0f;

    typename VecOf<TA>::T aReg[4];
    float4 bReg[4];

    // Issue global loads for the K-tile at k0 into registers (no LDS touch).
    auto loadTiles = [&](int k0) {
#pragma unroll
        for (int it = 0; it < 4; ++it) {
            typename VecOf<TA>::T za = {};
            int gm = m0 + aRow[it];
            if (FULL || (gm < M && (k0 + aC4[it]) < K))
                za = *(const typename VecOf<TA>::T*)(A + (size_t)gm * K + k0 + aC4[it]);
            aReg[it] = za;
        }
#pragma unroll
        for (int it = 0; it < 4; ++it) {
            float4 zb = {};
            int gn = n0 + bC4[it];
            int gk = k0 + bKr[it];
            if (FULL) {
                zb = *(const float4*)(B + (size_t)gk * N + gn);
            } else if (gk < K) {
                const float* bp = B + (size_t)gk * N + gn;
                if (gn + 3 < N) {
                    zb = *(const float4*)bp;
                } else {
#pragma unroll
                    for (int i = 0; i < 4; ++i)
                        if (gn + i < N) ((float*)&zb)[i] = bp[i];
                }
            }
            bReg[it] = zb;
        }
        // Warm L2 one more tile ahead (global_prefetch_b8).
        int kpf = k0 + BK;
        if (kpf < K) {
            __builtin_prefetch(A + (size_t)(m0 + aRow[0]) * K + kpf + aC4[0], 0, 1);
            __builtin_prefetch(B + (size_t)(kpf + bKr[0]) * N + n0 + bC4[0], 0, 1);
        }
    };

    // Convert staged registers to bf16 and deposit into LDS.
    auto storeTiles = [&]() {
#pragma unroll
        for (int it = 0; it < 4; ++it)
            stA(&As[aRow[it]][aC4[it]], aReg[it]);
#pragma unroll
        for (int it = 0; it < 4; ++it) {
            Bs[bC4[it] + 0][bKr[it]] = f2bf_fast(bReg[it].x);
            Bs[bC4[it] + 1][bKr[it]] = f2bf_fast(bReg[it].y);
            Bs[bC4[it] + 2][bKr[it]] = f2bf_fast(bReg[it].z);
            Bs[bC4[it] + 3][bKr[it]] = f2bf_fast(bReg[it].w);
        }
    };

    const int nTiles = (K + BK - 1) / BK;
    loadTiles(0);

    for (int t = 0; t < nTiles; ++t) {
        storeTiles();                     // waits on in-flight loads here
        __syncthreads();
        if (t + 1 < nTiles) loadTiles((t + 1) * BK);   // overlap with WMMAs

        // ---- WMMA compute: 4 A-frags x 2 B-frags ----
        bf16x16 af[4];
#pragma unroll
        for (int mt = 0; mt < 4; ++mt) {
            const unsigned short* p = &As[wr * 64 + mt * 16 + lmod][kA];
            FragU f;
            f.u[0] = *(const uint4*)(p);        // K kA .. kA+7
            f.u[1] = *(const uint4*)(p + 16);   // K kA+16 .. kA+23
            af[mt] = f.v;
        }
#pragma unroll
        for (int nt = 0; nt < 2; ++nt) {
            const unsigned short* p = &Bs[wc * 32 + nt * 16 + lmod][kB];
            FragU f;
            f.u[0] = *(const uint4*)(p);        // K kB .. kB+7
            f.u[1] = *(const uint4*)(p + 16);   // K kB+16 .. kB+23
            bf16x16 bfrag = f.v;
#pragma unroll
            for (int mt = 0; mt < 4; ++mt) {
                acc[mt][nt] = __builtin_amdgcn_wmma_f32_16x16x32_bf16(
                    false, af[mt], false, bfrag, (short)0, acc[mt][nt],
                    false, false);
            }
        }
        __syncthreads();
    }

    // ---- store C: VGPR g of lanes 0-15 -> row g, lanes 16-31 -> row 8+g ----
    const int mrow0 = m0 + wr * 64;
#pragma unroll
    for (int mt = 0; mt < 4; ++mt)
#pragma unroll
        for (int nt = 0; nt < 2; ++nt) {
            int c = n0 + wc * 32 + nt * 16 + lmod;
#pragma unroll
            for (int g = 0; g < 8; ++g) {
                int r = mrow0 + mt * 16 + laneHalf * 8 + g;
                if (FULL || (r < M && c < N))
                    C[(size_t)r * ldc + c] = acc[mt][nt][g];
            }
        }
}

// ---------------------------------------------------------------------------
// rmsnorm of cq rows (C=1536), output as bf16 payloads for GEMM2's A operand.
// ---------------------------------------------------------------------------
__global__ __launch_bounds__(256)
void rmsnorm_cq_kernel(const float* __restrict__ cq, const float* __restrict__ gamma,
                       unsigned short* __restrict__ cqn, int C)
{
    __shared__ float red[256];
    const int row = blockIdx.x;
    const float* src = cq + (size_t)row * C;
    float ss = 0.f;
    for (int i = threadIdx.x; i < C; i += 256) { float v = src[i]; ss += v * v; }
    red[threadIdx.x] = ss;
    __syncthreads();
    for (int s = 128; s > 0; s >>= 1) {
        if (threadIdx.x < s) red[threadIdx.x] += red[threadIdx.x + s];
        __syncthreads();
    }
    float scale = rsqrtf(red[0] / (float)C + EPSF);
    unsigned short* dst = cqn + (size_t)row * C;
    for (int i = threadIdx.x; i < C; i += 256)
        dst[i] = f2bf_fast(src[i] * scale * gamma[i]);
}

// ---------------------------------------------------------------------------
// KV epilogue: ckv = rmsnorm(kv[:, :512]) -> out cols [24576, 25088);
//              k_rope = rope(kv[:, 512:576]) -> out cols [25088, 25152).
// ---------------------------------------------------------------------------
__global__ __launch_bounds__(256)
void kv_epilogue_kernel(const float* __restrict__ kv, const float* __restrict__ gamma,
                        const float* __restrict__ rc, const float* __restrict__ rs,
                        float* __restrict__ out)
{
    __shared__ float red[256];
    const int row = blockIdx.x;
    const float* src = kv + (size_t)row * KV_COLS;
    float ss = 0.f;
    for (int i = threadIdx.x; i < KV_RANK; i += 256) { float v = src[i]; ss += v * v; }
    red[threadIdx.x] = ss;
    __syncthreads();
    for (int s = 128; s > 0; s >>= 1) {
        if (threadIdx.x < s) red[threadIdx.x] += red[threadIdx.x + s];
        __syncthreads();
    }
    float scale = rsqrtf(red[0] / (float)KV_RANK + EPSF);
    float* dst = out + (size_t)row * OUT_C + Q_COLS;
    for (int i = threadIdx.x; i < KV_RANK; i += 256)
        dst[i] = src[i] * scale * gamma[i];

    if (threadIdx.x < QK_ROPE / 2) {
        int j = threadIdx.x;
        const float* cr = rc + (size_t)row * QK_ROPE;
        const float* sr = rs + (size_t)row * QK_ROPE;
        float x1 = src[KV_RANK + j];
        float x2 = src[KV_RANK + 32 + j];
        float* rdst = out + (size_t)row * OUT_C + Q_COLS + KV_RANK;
        rdst[j]      = x1 * cr[j]      - x2 * sr[j];        // x*cos + (-x2)*sin
        rdst[32 + j] = x2 * cr[32 + j] + x1 * sr[32 + j];   // x*cos + ( x1)*sin
    }
}

// ---------------------------------------------------------------------------
// In-place RoPE on q: per (t, head, j<32) rotate cols [h*192+128+j, +32].
// Each thread owns both halves of a pair -> race-free, deterministic.
// ---------------------------------------------------------------------------
__global__ __launch_bounds__(256)
void q_rope_kernel(float* __restrict__ out, const float* __restrict__ rc,
                   const float* __restrict__ rs, int total)
{
    int idx = blockIdx.x * 256 + threadIdx.x;
    if (idx >= total) return;
    int j = idx & 31;
    int h = (idx >> 5) & (N_HEADS - 1);
    int t = idx >> 12;                     // / (128 heads * 32 pairs)
    float* p = out + (size_t)t * OUT_C + h * (QK_NOPE + QK_ROPE) + QK_NOPE;
    const float* cr = rc + (size_t)t * QK_ROPE;
    const float* sr = rs + (size_t)t * QK_ROPE;
    float x1 = p[j];
    float x2 = p[j + 32];
    p[j]      = x1 * cr[j]      - x2 * sr[j];
    p[j + 32] = x2 * cr[32 + j] + x1 * sr[32 + j];
}

// ---------------------------------------------------------------------------
template <typename TA>
static inline void launch_gemm(const TA* A, const float* B, float* C,
                               int M, int N, int K, int ldc, hipStream_t stream)
{
    dim3 blk(256);
    dim3 grd((N + BN - 1) / BN, (M + BM - 1) / BM);
    bool full = (M % BM == 0) && (N % BN == 0) && (K % BK == 0);
    if (full)
        gemm_bf16_wmma<TA, true ><<<grd, blk, 0, stream>>>(A, B, C, M, N, K, ldc);
    else
        gemm_bf16_wmma<TA, false><<<grd, blk, 0, stream>>>(A, B, C, M, N, K, ldc);
}

extern "C" void kernel_launch(void* const* d_in, const int* in_sizes, int n_in,
                              void* d_out, int out_size, void* d_ws, size_t ws_size,
                              hipStream_t stream)
{
    const float* token_x  = (const float*)d_in[0];
    const float* wq_a     = (const float*)d_in[1];
    const float* wq_b     = (const float*)d_in[2];
    const float* wkv      = (const float*)d_in[3];
    const float* rope_cos = (const float*)d_in[4];
    const float* rope_sin = (const float*)d_in[5];
    const float* gamma_cq = (const float*)d_in[6];
    const float* gamma_ck = (const float*)d_in[7];
    float* out = (float*)d_out;

    const int T = in_sizes[4] / QK_ROPE;       // 4096
    const int H = in_sizes[0] / T;             // 7168

    // Workspace layout (fp32 cq, fp32 kv, bf16 cq_norm)
    char* ws = (char*)d_ws;
    size_t cq_bytes  = (size_t)T * CQ_DIM * sizeof(float);
    size_t kv_bytes  = (size_t)T * KV_COLS * sizeof(float);
    float*          cq  = (float*)ws;
    float*          kv  = (float*)(ws + cq_bytes);
    unsigned short* cqn = (unsigned short*)(ws + cq_bytes + kv_bytes);

    // 1) cq_raw = token_x @ wq_a     (4096 x 7168 x 1536)
    launch_gemm<float>(token_x, wq_a, cq, T, CQ_DIM, H, CQ_DIM, stream);
    // 2) kv = token_x @ wkv          (4096 x 7168 x 576)
    launch_gemm<float>(token_x, wkv, kv, T, KV_COLS, H, KV_COLS, stream);
    // 3) cq_norm = rmsnorm(cq) -> bf16
    rmsnorm_cq_kernel<<<T, 256, 0, stream>>>(cq, gamma_cq, cqn, CQ_DIM);
    // 4) q = cq_norm @ wq_b -> out[:, :24576]  (4096 x 1536 x 24576)
    launch_gemm<unsigned short>(cqn, wq_b, out, T, Q_COLS, CQ_DIM, OUT_C, stream);
    // 5) rope on q rope-dims, in place
    int total = T * N_HEADS * (QK_ROPE / 2);
    q_rope_kernel<<<(total + 255) / 256, 256, 0, stream>>>(out, rope_cos, rope_sin, total);
    // 6) ckv rmsnorm + k_rope -> out[:, 24576:]
    kv_epilogue_kernel<<<T, 256, 0, stream>>>(kv, gamma_ck, rope_cos, rope_sin, out);
}
